// Decoder_72679436583460
// MI455X (gfx1250) — compile-verified, matrix operand order checked by reading, not run on hardware
//
#include <hip/hip_runtime.h>
#include <hip/hip_bf16.h>
#include <math.h>
#include <stdint.h>

// ---------------------------------------------------------------------------
// Decoder (GRU + dot-attention + classifier), teacher forcing.
// V=32000 E=512 H=1024 B=16 S=128 T=128.
//
// MI455X strategy:
//  * Recurrence: 2 kernels/step. GRU gates via v_wmma_f32_16x16x32_bf16
//    (M=16 = one tile row); attention kernel also prepares next step's
//    bf16 inputs (embed gather + ctx concat + h conversion) -> no extra kernel.
//  * Classifier hoisted out of the time loop into ONE big bf16 WMMA GEMM
//    [2048,2048]x[2048,32000]: clf_W converted once to bf16 (131MB, resident
//    in the 192MB L2), staged to LDS with global_load_async_to_lds_b128
//    (ASYNCcnt-tracked, double buffered), fp32 accumulate, fused bias +
//    [B,T,V] scatter.
// ---------------------------------------------------------------------------

typedef __attribute__((ext_vector_type(16))) __bf16 v16bf;
typedef __attribute__((ext_vector_type(8)))  __bf16 v8bf;
typedef __attribute__((ext_vector_type(8)))  float  v8f;

#define Bsz 16
#define Ssz 128
#define Tsz 128
#define Hsz 1024
#define Esz 512
#define Vsz 32000
#define KIH 1536   // E + H
#define K3H 3072   // 3*H
#define KP  2048   // 2*H (pred width / GEMM K)

// --- WMMA helpers ----------------------------------------------------------

__device__ __forceinline__ v8f wmma_bf16(v16bf a, v16bf b, v8f c) {
  return __builtin_amdgcn_wmma_f32_16x16x32_bf16(
      false, a, false, b, (short)0, c, false, false);
}

// 16-bit A/B fragment per ISA layout: lane chunk0 = k in [8*half, +8),
// chunk1 = k in [16+8*half, +8); rowk points at element [row][kb].
__device__ __forceinline__ v16bf load_frag(const __bf16* rowk, int half) {
  const __bf16* p = rowk + (half << 3);
  union { v16bf v; v8bf h[2]; } u;
  u.h[0] = *reinterpret_cast<const v8bf*>(p);
  u.h[1] = *reinterpret_cast<const v8bf*>(p + 16);
  return u.v;
}

__device__ __forceinline__ float sigm(float x) { return 1.0f / (1.0f + expf(-x)); }

// --- 0) fp32 -> bf16 weight conversion -------------------------------------

__global__ void k_f32_to_bf16(const float* __restrict__ src,
                              unsigned short* __restrict__ dst_, size_t n) {
  __bf16* dst = reinterpret_cast<__bf16*>(dst_);
  size_t i = (size_t)blockIdx.x * blockDim.x + threadIdx.x;
  size_t stride = (size_t)gridDim.x * blockDim.x;
  for (; i < n; i += stride) dst[i] = (__bf16)src[i];
}

// --- 1) h0 = init_h @ hm_W^T + hm_b ; build X(t=0), Hb(t=0) ------------------
// 24576 threads: [0,16384) -> h0/Hb/X-ctx(=0); [16384,24576) -> X-emb(y_0).

__global__ void k_init(const float* __restrict__ ih, const float* __restrict__ hmW,
                       const float* __restrict__ hmb, const int* __restrict__ tgt,
                       const float* __restrict__ embW, float* __restrict__ hf,
                       unsigned short* __restrict__ X_, unsigned short* __restrict__ Hb_) {
  __bf16* X  = reinterpret_cast<__bf16*>(X_);
  __bf16* Hb = reinterpret_cast<__bf16*>(Hb_);
  int tid = blockIdx.x * blockDim.x + threadIdx.x;
  if (tid < Bsz * Hsz) {
    int b = tid >> 10, j = tid & (Hsz - 1);
    const float* x = ih + (size_t)b * Hsz;
    const float* w = hmW + (size_t)j * Hsz;
    float a = hmb[j];
    for (int k = 0; k < Hsz; ++k) a += x[k] * w[k];
    hf[tid] = a;
    Hb[tid] = (__bf16)a;
    X[b * KIH + Esz + j] = (__bf16)0.0f;       // ctx0 = 0
  } else if (tid < Bsz * Hsz + Bsz * Esz) {
    int i2 = tid - Bsz * Hsz;
    int b = i2 >> 9, c = i2 & (Esz - 1);
    int y = tgt[b * Tsz + 0];
    X[b * KIH + c] = (__bf16)embW[(size_t)y * Esz + c];
  }
}

// --- 2) per-step GRU cell: WMMA gates + fused update ------------------------
// 64 workgroups x 1 wave; workgroup jt owns output columns [16*jt, 16*jt+16).

__global__ __launch_bounds__(32) void k_gru(
    const unsigned short* __restrict__ X_, const unsigned short* __restrict__ Hb_,
    const unsigned short* __restrict__ wih_, const unsigned short* __restrict__ whh_,
    const float* __restrict__ bih, const float* __restrict__ bhh,
    float* __restrict__ hf, unsigned short* __restrict__ pred_, int t) {
  const __bf16* X   = reinterpret_cast<const __bf16*>(X_);
  const __bf16* Hb  = reinterpret_cast<const __bf16*>(Hb_);
  const __bf16* wih = reinterpret_cast<const __bf16*>(wih_);
  const __bf16* whh = reinterpret_cast<const __bf16*>(whh_);
  __bf16* pred = reinterpret_cast<__bf16*>(pred_);

  int lane = threadIdx.x;
  int half = lane >> 4, ln = lane & 15;
  int j0 = blockIdx.x << 4;
  const v8f z8 = {0, 0, 0, 0, 0, 0, 0, 0};
  v8f a_ir = z8, a_iz = z8, a_in = z8, a_hr = z8, a_hz = z8, a_hn = z8;

  // gi = x @ W_ih^T : A = X[16x1536], B cols = W_ih rows (r/z/n gates)
  const __bf16* wr = wih + (size_t)(j0 + ln) * KIH;
  const __bf16* wz = wih + (size_t)(Hsz + j0 + ln) * KIH;
  const __bf16* wn = wih + (size_t)(2 * Hsz + j0 + ln) * KIH;
  const __bf16* xr = X + (size_t)ln * KIH;
  for (int kb = 0; kb < KIH; kb += 32) {
    v16bf af = load_frag(xr + kb, half);
    a_ir = wmma_bf16(af, load_frag(wr + kb, half), a_ir);
    a_iz = wmma_bf16(af, load_frag(wz + kb, half), a_iz);
    a_in = wmma_bf16(af, load_frag(wn + kb, half), a_in);
  }
  // gh = h @ W_hh^T
  const __bf16* vr = whh + (size_t)(j0 + ln) * Hsz;
  const __bf16* vz = whh + (size_t)(Hsz + j0 + ln) * Hsz;
  const __bf16* vn = whh + (size_t)(2 * Hsz + j0 + ln) * Hsz;
  const __bf16* hr0 = Hb + (size_t)ln * Hsz;
  for (int kb = 0; kb < Hsz; kb += 32) {
    v16bf af = load_frag(hr0 + kb, half);
    a_hr = wmma_bf16(af, load_frag(vr + kb, half), a_hr);
    a_hz = wmma_bf16(af, load_frag(vz + kb, half), a_hz);
    a_hn = wmma_bf16(af, load_frag(vn + kb, half), a_hn);
  }

  int col = j0 + ln;
  float bir = bih[col], biz = bih[Hsz + col], bin = bih[2 * Hsz + col];
  float bhr = bhh[col], bhz = bhh[Hsz + col], bhn = bhh[2 * Hsz + col];
#pragma unroll
  for (int r = 0; r < 8; ++r) {
    int m = r + (half << 3);                          // batch row
    float rr = sigm((a_ir[r] + bir) + (a_hr[r] + bhr));
    float zz = sigm((a_iz[r] + biz) + (a_hz[r] + bhz));
    float nn = tanhf((a_in[r] + bin) + rr * (a_hn[r] + bhn));
    float ho = hf[m * Hsz + col];
    float hn2 = (1.0f - zz) * nn + zz * ho;
    hf[m * Hsz + col] = hn2;
    pred[((size_t)t * Bsz + m) * KP + Hsz + col] = (__bf16)hn2;   // h half of pred
  }
}

// --- 3) per-step attention + next-step input prep ---------------------------
// One workgroup (128 threads) per batch element. After softmax/ctx it writes
// pred[ctx half], X(t+1) = [emb(y_{t+1}), ctx], Hb(t+1) = bf16(h_new).

__global__ __launch_bounds__(128) void k_attn(
    const float* __restrict__ enc, const float* __restrict__ hf,
    const int* __restrict__ tgt, const float* __restrict__ embW,
    unsigned short* __restrict__ pred_, unsigned short* __restrict__ X_,
    unsigned short* __restrict__ Hb_, int t) {
  __bf16* pred = reinterpret_cast<__bf16*>(pred_);
  __bf16* X  = reinterpret_cast<__bf16*>(X_);
  __bf16* Hb = reinterpret_cast<__bf16*>(Hb_);
  __shared__ float sh[Hsz];
  __shared__ float sp[Ssz];
  __shared__ float red[Ssz];
  int b = blockIdx.x, s = threadIdx.x;
  for (int i = s; i < Hsz; i += Ssz) sh[i] = hf[b * Hsz + i];
  __syncthreads();
  const float* er = enc + ((size_t)b * Ssz + s) * Hsz;
  float acc = 0.0f;
  for (int k = 0; k < Hsz; ++k) acc += er[k] * sh[k];
  red[s] = acc;
  __syncthreads();
  for (int off = Ssz / 2; off > 0; off >>= 1) {
    if (s < off) red[s] = fmaxf(red[s], red[s + off]);
    __syncthreads();
  }
  float mx = red[0];
  __syncthreads();
  float e = expf(acc - mx);
  red[s] = e;
  __syncthreads();
  for (int off = Ssz / 2; off > 0; off >>= 1) {
    if (s < off) red[s] += red[s + off];
    __syncthreads();
  }
  float inv = 1.0f / red[0];
  sp[s] = e * inv;
  __syncthreads();

  const bool more = (t + 1) < Tsz;
  for (int c = s; c < Hsz; c += Ssz) {
    float a2 = 0.0f;
    for (int s2 = 0; s2 < Ssz; ++s2)
      a2 += enc[((size_t)b * Ssz + s2) * Hsz + c] * sp[s2];
    __bf16 cb = (__bf16)a2;
    pred[((size_t)t * Bsz + b) * KP + c] = cb;       // ctx half of pred
    if (more) {
      X[b * KIH + Esz + c] = cb;                     // ctx part of next x
      Hb[b * Hsz + c] = (__bf16)sh[c];               // bf16(h_new)
    }
  }
  if (more) {
    int y = tgt[b * Tsz + (t + 1)];
    for (int c = s; c < Esz; c += Ssz)
      X[b * KIH + c] = (__bf16)embW[(size_t)y * Esz + c];
  }
}

// --- 4) big classifier GEMM: [2048,2048] x [32000,2048]^T -> out[B,T,V] -----
// 128x128x32 blocks, 8 waves (4Mx2N), each wave 2x4 tiles. Panels staged to
// LDS with global_load_async_to_lds_b128 (ASYNCcnt), double buffered.

#define LDK 40   // padded LDS row stride (bf16) -> conflict-free ds_load_b128

__global__ __launch_bounds__(256) void k_gemm(
    const unsigned short* __restrict__ pred_, const unsigned short* __restrict__ clfW_,
    const float* __restrict__ clfb, float* __restrict__ out) {
  const __bf16* pred = reinterpret_cast<const __bf16*>(pred_);
  const __bf16* clfW = reinterpret_cast<const __bf16*>(clfW_);
  __shared__ __bf16 As[2][128][LDK];
  __shared__ __bf16 Bs[2][128][LDK];

  int tid = threadIdx.x;
  int lane = tid & 31, wid = tid >> 5;
  int half = lane >> 4, ln = lane & 15;
  int wm = wid & 3, wn = wid >> 2;          // 4 waves in M, 2 in N
  int M0 = blockIdx.y * 128, N0 = blockIdx.x * 128;

  // Each thread stages one 32B slice (two b128) per matrix per panel:
  // row = tid>>1, elements [16*(tid&1), 16*(tid&1)+16).
  int lrow = tid >> 1, lk = (tid & 1) * 16;
  const __bf16* gA = pred + (size_t)(M0 + lrow) * KP + lk;
  const __bf16* gB = clfW + (size_t)(N0 + lrow) * KP + lk;

  const v8f z8 = {0, 0, 0, 0, 0, 0, 0, 0};
  v8f acc[2][4];
#pragma unroll
  for (int i = 0; i < 2; ++i)
#pragma unroll
    for (int j = 0; j < 4; ++j) acc[i][j] = z8;

  auto stage = [&](int p, int kb) {
    uint32_t la = (uint32_t)(uintptr_t)&As[p][lrow][lk];
    uint32_t lb = (uint32_t)(uintptr_t)&Bs[p][lrow][lk];
    uint64_t ga = (uint64_t)(uintptr_t)(gA + kb);
    uint64_t gb = (uint64_t)(uintptr_t)(gB + kb);
    asm volatile("global_load_async_to_lds_b128 %0, %1, off"
                 :: "v"(la), "v"(ga) : "memory");
    asm volatile("global_load_async_to_lds_b128 %0, %1, off offset:16"
                 :: "v"(la), "v"(ga) : "memory");
    asm volatile("global_load_async_to_lds_b128 %0, %1, off"
                 :: "v"(lb), "v"(gb) : "memory");
    asm volatile("global_load_async_to_lds_b128 %0, %1, off offset:16"
                 :: "v"(lb), "v"(gb) : "memory");
  };

  stage(0, 0);
  const int NK = KP / 32;   // 64
  for (int k = 0; k < NK; ++k) {
    int p = k & 1;
    asm volatile("s_wait_asynccnt 0x0" ::: "memory");  // my panel-p ops landed
    __syncthreads();                                   // everyone's landed; also
                                                       // orders prior reads vs
                                                       // this iter's overwrite
    v16bf af[2], bf[4];
#pragma unroll
    for (int tm = 0; tm < 2; ++tm)
      af[tm] = load_frag(&As[p][wm * 32 + tm * 16 + ln][0], half);
#pragma unroll
    for (int tn = 0; tn < 4; ++tn)
      bf[tn] = load_frag(&Bs[p][wn * 64 + tn * 16 + ln][0], half);
#pragma unroll
    for (int tm = 0; tm < 2; ++tm)
#pragma unroll
      for (int tn = 0; tn < 4; ++tn)
        acc[tm][tn] = wmma_bf16(af[tm], bf[tn], acc[tm][tn]);
    if (k + 1 < NK) stage(p ^ 1, (k + 1) * 32);        // overlap with wmma
  }

  // epilogue: bias + scatter to out[b][t][v]  (GEMM row R = t*16 + b)
#pragma unroll
  for (int tm = 0; tm < 2; ++tm) {
#pragma unroll
    for (int tn = 0; tn < 4; ++tn) {
      int C = N0 + wn * 64 + tn * 16 + ln;
      float bias = clfb[C];
#pragma unroll
      for (int r = 0; r < 8; ++r) {
        int R = M0 + wm * 32 + tm * 16 + r + (half << 3);
        int tt = R >> 4, bb = R & 15;
        out[((size_t)bb * Tsz + tt) * Vsz + C] = acc[tm][tn][r] + bias;
      }
    }
  }
}

// ---------------------------------------------------------------------------

extern "C" void kernel_launch(void* const* d_in, const int* in_sizes, int n_in,
                              void* d_out, int out_size, void* d_ws, size_t ws_size,
                              hipStream_t stream) {
  const float* enc  = (const float*)d_in[0];   // [16,128,1024]
  const float* ih0  = (const float*)d_in[1];   // [16,1024]
  const int*   tgt  = (const int*)d_in[2];     // [16,128]
  const float* embW = (const float*)d_in[3];   // [32000,512]
  const float* wih  = (const float*)d_in[4];   // [3072,1536]
  const float* whh  = (const float*)d_in[5];   // [3072,1024]
  const float* bih  = (const float*)d_in[6];   // [3072]
  const float* bhh  = (const float*)d_in[7];   // [3072]
  const float* hmW  = (const float*)d_in[8];   // [1024,1024]
  const float* hmb  = (const float*)d_in[9];   // [1024]
  const float* clfW = (const float*)d_in[10];  // [32000,2048]
  const float* clfb = (const float*)d_in[11];  // [32000]
  float* out = (float*)d_out;                  // [16,128,32000]

  char* ws = (char*)d_ws;
  size_t off = 0;
  auto carve = [&](size_t bytes) {
    size_t o = off;
    off = (off + bytes + 255) & ~(size_t)255;
    return o;
  };
  const size_t nCLF = (size_t)Vsz * KP;
  const size_t nWIH = (size_t)K3H * KIH;
  const size_t nWHH = (size_t)K3H * Hsz;
  const size_t nPRED = (size_t)Tsz * Bsz * KP;
  unsigned short* clfWb = (unsigned short*)(ws + carve(nCLF * 2));
  unsigned short* wihb  = (unsigned short*)(ws + carve(nWIH * 2));
  unsigned short* whhb  = (unsigned short*)(ws + carve(nWHH * 2));
  unsigned short* predb = (unsigned short*)(ws + carve(nPRED * 2));
  unsigned short* Xb    = (unsigned short*)(ws + carve((size_t)Bsz * KIH * 2));
  unsigned short* Hbb   = (unsigned short*)(ws + carve((size_t)Bsz * Hsz * 2));
  float*          hf    = (float*)(ws + carve((size_t)Bsz * Hsz * 4));
  (void)ws_size; (void)in_sizes; (void)n_in; (void)out_size;

  // Weights -> bf16 (clf_W bf16 = 131MB, L2-resident on the 192MB L2).
  k_f32_to_bf16<<<2048, 256, 0, stream>>>(clfW, clfWb, nCLF);
  k_f32_to_bf16<<<1024, 256, 0, stream>>>(wih, wihb, nWIH);
  k_f32_to_bf16<<<1024, 256, 0, stream>>>(whh, whhb, nWHH);

  // h0 = hidden_map(init_h); build X/Hb for t=0 (ctx0 = 0).
  k_init<<<96, 256, 0, stream>>>(ih0, hmW, hmb, tgt, embW, hf, Xb, Hbb);

  // Sequential recurrence: 2 kernels per step (gates+update, attn+next-prep).
  for (int t = 0; t < Tsz; ++t) {
    k_gru<<<64, 32, 0, stream>>>(Xb, Hbb, wihb, whhb, bih, bhh, hf, predb, t);
    k_attn<<<16, 128, 0, stream>>>(enc, hf, tgt, embW, predb, Xb, Hbb, t);
  }

  // One big WMMA GEMM for all logits: [2048,2048] x [2048,32000] + bias.
  k_gemm<<<dim3(Vsz / 128, (Tsz * Bsz) / 128), 256, 0, stream>>>(predb, clfWb, clfb, out);
}
